// MultiHeadAttention_7146825581216
// MI455X (gfx1250) — matrix operand order repaired and checked by reading, not compiled, not run on hardware
//
#include <hip/hip_runtime.h>
#include <hip/hip_bf16.h>

typedef __attribute__((ext_vector_type(16))) _Float16 v16h;
typedef __attribute__((ext_vector_type(8)))  float    v8f;

#define BATCH 4
#define SEQ   2048
#define DMODEL 1024
#define NH    16
#define HDIM  64

__device__ inline v8f wmma_f16(v16h a, v16h b, v8f c) {
  return __builtin_amdgcn_wmma_f32_16x16x32_f16(
      false, a, false, b, (short)0, c, false, false);
}

// A-fragment (16x32, f16 source, row-major, row stride ld).
// ISA layout: lane<16 -> M=lane, K = 0..7 and 16..23 ; lane>=16 -> M=lane-16, K = 8..15 and 24..31
__device__ inline v16h load_a_f16(const _Float16* src, int ld, int lane) {
  int m  = lane & 15;
  int kh = (lane >> 4) & 1;
  const _Float16* p = src + m * ld + kh * 8;
  v16h a;
#pragma unroll
  for (int e = 0; e < 8; ++e) a[e] = p[e];
#pragma unroll
  for (int e = 0; e < 8; ++e) a[8 + e] = p[16 + e];
  return a;
}

// A-fragment from f32 source (convert in-register)
__device__ inline v16h load_a_f32(const float* src, int ld, int lane) {
  int m  = lane & 15;
  int kh = (lane >> 4) & 1;
  const float* p = src + m * ld + kh * 8;
  v16h a;
#pragma unroll
  for (int e = 0; e < 8; ++e) a[e] = (_Float16)p[e];
#pragma unroll
  for (int e = 0; e < 8; ++e) a[8 + e] = (_Float16)p[16 + e];
  return a;
}

// B-fragment (32x16) from N-major storage: src[n*ld + k].
// ISA layout: lane = N column (lane&15), lane<16 -> K = 0..7 and 16..23 ; lane>=16 -> K = 8..15 and 24..31
__device__ inline v16h load_b_nmajor(const _Float16* src, int ld, int lane) {
  int n  = lane & 15;
  int kh = (lane >> 4) & 1;
  const _Float16* p = src + (size_t)n * ld + kh * 8;
  v16h b;
#pragma unroll
  for (int e = 0; e < 8; ++e) b[e] = p[e];
#pragma unroll
  for (int e = 0; e < 8; ++e) b[8 + e] = p[16 + e];
  return b;
}

// ---------------- Kernel 1: weight convert + transpose to f16 N-major ----------------
__global__ void wconv_kernel(const float* __restrict__ Wq, const float* __restrict__ Wk,
                             const float* __restrict__ Wv, const float* __restrict__ Wo,
                             _Float16* __restrict__ Wqt, _Float16* __restrict__ Wkt,
                             _Float16* __restrict__ Wvt, _Float16* __restrict__ Wot) {
  int idx = blockIdx.x * 256 + threadIdx.x;       // 0 .. 1M-1
  const float* W; _Float16* Wt;
  switch (blockIdx.y) {
    case 0: W = Wq; Wt = Wqt; break;
    case 1: W = Wk; Wt = Wkt; break;
    case 2: W = Wv; Wt = Wvt; break;
    default: W = Wo; Wt = Wot; break;
  }
  int n = idx >> 10;        // output column of original matrix
  int k = idx & 1023;       // K dim
  Wt[idx] = (_Float16)W[k * 1024 + n];   // Wt[n][k] = W[k][n]
}

// ------- Kernel 2: fused QKV projection GEMM, 16(M) x 64(N) per wave (A reused 4x) ---
// N quad of 64 == exactly one head h = nq for the [B,H,S,HD] stores.
__global__ void qkv_proj_kernel(const float* __restrict__ inq, const float* __restrict__ ink,
                                const float* __restrict__ inv_,
                                const _Float16* __restrict__ Wqt, const _Float16* __restrict__ Wkt,
                                const _Float16* __restrict__ Wvt,
                                const float* __restrict__ bq, const float* __restrict__ bk,
                                const float* __restrict__ bv,
                                _Float16* __restrict__ Qh, _Float16* __restrict__ Kh,
                                _Float16* __restrict__ Vth) {
  int mode = blockIdx.y;
  const float* A; const _Float16* Bt; const float* bias;
  if (mode == 0)      { A = inq;  Bt = Wqt; bias = bq; }
  else if (mode == 1) { A = ink;  Bt = Wkt; bias = bk; }
  else                { A = inv_; Bt = Wvt; bias = bv; }

  int tile = blockIdx.x;          // 512 Mtiles x 16 Nquads
  int nq = tile & 15;             // head index
  int mt = tile >> 4;
  int lane = threadIdx.x;

  const float*    Arow = A  + (size_t)mt * 16 * DMODEL;
  const _Float16* Btt  = Bt + (size_t)nq * 64 * DMODEL;

  v8f c[4] = {};
#pragma unroll 2
  for (int kb = 0; kb < DMODEL; kb += 32) {
    v16h a = load_a_f32(Arow + kb, DMODEL, lane);
#pragma unroll
    for (int nt = 0; nt < 4; ++nt) {
      v16h b = load_b_nmajor(Btt + (size_t)nt * 16 * DMODEL + kb, DMODEL, lane);
      c[nt] = wmma_f16(a, b, c[nt]);
    }
  }

  int hi = lane >> 4;
  int nl = lane & 15;
  int h  = nq;                               // head for this N quad

#pragma unroll
  for (int nt = 0; nt < 4; ++nt) {
    int hd   = nt * 16 + nl;
    float bval = bias[h * HDIM + hd];
#pragma unroll
    for (int r = 0; r < 8; ++r) {
      int row = mt * 16 + r + 8 * hi;        // global row = b*SEQ + s
      int bb  = row >> 11;
      int s   = row & (SEQ - 1);
      float val = c[nt][r] + bval;
      if (mode == 0) {
        // Q pre-scaled by 1/sqrt(HD), layout [B,H,S,HD]
        Qh[((((size_t)bb * NH + h) * SEQ + s) * HDIM) + hd] = (_Float16)(val * 0.125f);
      } else if (mode == 1) {
        Kh[((((size_t)bb * NH + h) * SEQ + s) * HDIM) + hd] = (_Float16)val;
      } else {
        // V transposed: [B,H,HD,S]
        Vth[((((size_t)bb * NH + h) * HDIM + hd) * SEQ) + s] = (_Float16)val;
      }
    }
  }
}

// ---- Kernel 3: flash attention, one (b,h, 32-query group) per wave (K/V reused 2x) --
__global__ void attn_kernel(const _Float16* __restrict__ Qh, const _Float16* __restrict__ Kh,
                            const _Float16* __restrict__ Vth, _Float16* __restrict__ Xh) {
  __shared__ _Float16 Plds[2 * 16 * 32];   // P staging per query sub-tile

  int tile = blockIdx.x;               // 64 (b*H+h) x 64 query groups of 32
  int qg   = tile & 63;
  int bh   = tile >> 6;
  int lane = threadIdx.x;
  int hi   = lane >> 4;
  int nl   = lane & 15;

  const _Float16* Qt  = Qh  + (size_t)bh * SEQ * HDIM + (size_t)qg * 32 * HDIM;
  const _Float16* Kbh = Kh  + (size_t)bh * SEQ * HDIM;
  const _Float16* Vbh = Vth + (size_t)bh * HDIM * SEQ;

  // Q fragments are loop-invariant: hoist them into registers (2 sub-tiles x 2 K-steps)
  v16h qa[2][2];
#pragma unroll
  for (int qt = 0; qt < 2; ++qt)
#pragma unroll
    for (int kk = 0; kk < 2; ++kk)
      qa[qt][kk] = load_a_f16(Qt + (size_t)qt * 16 * HDIM + kk * 32, HDIM, lane);

  v8f oacc[2][4] = {};
  float mrow[2][8], lrow[2][8];
#pragma unroll
  for (int qt = 0; qt < 2; ++qt)
#pragma unroll
    for (int r = 0; r < 8; ++r) { mrow[qt][r] = -1e30f; lrow[qt][r] = 0.0f; }

  for (int kb = 0; kb < SEQ / 32; ++kb) {
    int keyBase = kb * 32;
    // prefetch next key block of K and V into the cache hierarchy
    if (kb + 1 < SEQ / 32) {
      __builtin_prefetch(Kbh + (size_t)(keyBase + 32) * HDIM + lane * 64, 0, 3);
      __builtin_prefetch(Vbh + (size_t)(lane * 2) * SEQ + keyBase + 32, 0, 3);
    }
    // --- scores: 2 x (16 queries x 32 keys), shared K fragments ---
    v8f s[2][2] = {};
#pragma unroll
    for (int kk = 0; kk < 2; ++kk) {
      v16h b0 = load_b_nmajor(Kbh + (size_t)keyBase * HDIM + kk * 32, HDIM, lane);
      v16h b1 = load_b_nmajor(Kbh + (size_t)(keyBase + 16) * HDIM + kk * 32, HDIM, lane);
#pragma unroll
      for (int qt = 0; qt < 2; ++qt) {
        s[qt][0] = wmma_f16(qa[qt][kk], b0, s[qt][0]);
        s[qt][1] = wmma_f16(qa[qt][kk], b1, s[qt][1]);
      }
    }
    // --- issue V fragment loads early so they overlap the softmax VALU work ---
    v16h vb[4];
#pragma unroll
    for (int ct = 0; ct < 4; ++ct)
      vb[ct] = load_b_nmajor(Vbh + (size_t)(ct * 16) * SEQ + keyBase, SEQ, lane);
    // --- online softmax per query sub-tile ---
#pragma unroll
    for (int qt = 0; qt < 2; ++qt) {
#pragma unroll
      for (int r = 0; r < 8; ++r) {
        float bm = fmaxf(s[qt][0][r], s[qt][1][r]);
        bm = fmaxf(bm, __shfl_xor(bm, 1, 32));
        bm = fmaxf(bm, __shfl_xor(bm, 2, 32));
        bm = fmaxf(bm, __shfl_xor(bm, 4, 32));
        bm = fmaxf(bm, __shfl_xor(bm, 8, 32));
        float mnew = fmaxf(mrow[qt][r], bm);
        float corr = __expf(mrow[qt][r] - mnew);
        float p0 = __expf(s[qt][0][r] - mnew);
        float p1 = __expf(s[qt][1][r] - mnew);
        float bs = p0 + p1;
        bs += __shfl_xor(bs, 1, 32);
        bs += __shfl_xor(bs, 2, 32);
        bs += __shfl_xor(bs, 4, 32);
        bs += __shfl_xor(bs, 8, 32);
        lrow[qt][r] = lrow[qt][r] * corr + bs;
        mrow[qt][r] = mnew;
        oacc[qt][0][r] *= corr; oacc[qt][1][r] *= corr;
        oacc[qt][2][r] *= corr; oacc[qt][3][r] *= corr;
        int row = r + 8 * hi;
        Plds[qt * 512 + row * 32 + nl]      = (_Float16)p0;
        Plds[qt * 512 + row * 32 + 16 + nl] = (_Float16)p1;
      }
    }
    __syncthreads();
    // --- O += P(16x32) @ V(32x64): shared V fragments across both query sub-tiles ---
    v16h ap0 = load_a_f16(Plds, 32, lane);
    v16h ap1 = load_a_f16(Plds + 512, 32, lane);
#pragma unroll
    for (int ct = 0; ct < 4; ++ct) {
      oacc[0][ct] = wmma_f16(ap0, vb[ct], oacc[0][ct]);
      oacc[1][ct] = wmma_f16(ap1, vb[ct], oacc[1][ct]);
    }
    __syncthreads();
  }

  // --- normalize and store X as [B,S,H,HD] f16 ---
  int bb = bh >> 4;
  int h  = bh & 15;
#pragma unroll
  for (int qt = 0; qt < 2; ++qt) {
#pragma unroll
    for (int r = 0; r < 8; ++r) {
      float inv = 1.0f / lrow[qt][r];
      int s = qg * 32 + qt * 16 + r + 8 * hi;
      size_t base = (((size_t)bb * SEQ + s) * NH + h) * HDIM;
#pragma unroll
      for (int ct = 0; ct < 4; ++ct)
        Xh[base + ct * 16 + nl] = (_Float16)(oacc[qt][ct][r] * inv);
    }
  }
}

// -------- Kernel 4: output projection GEMM, 16(M) x 64(N) per wave, f32 store --------
__global__ void outproj_kernel(const _Float16* __restrict__ Xh, const _Float16* __restrict__ Wot,
                               const float* __restrict__ bo, float* __restrict__ out) {
  int tile = blockIdx.x;          // 512 Mtiles x 16 Nquads
  int nq = tile & 15;
  int mt = tile >> 4;
  int lane = threadIdx.x;

  const _Float16* Arow = Xh  + (size_t)mt * 16 * DMODEL;
  const _Float16* Btt  = Wot + (size_t)nq * 64 * DMODEL;

  v8f c[4] = {};
#pragma unroll 2
  for (int kb = 0; kb < DMODEL; kb += 32) {
    v16h a = load_a_f16(Arow + kb, DMODEL, lane);
#pragma unroll
    for (int nt = 0; nt < 4; ++nt) {
      v16h b = load_b_nmajor(Btt + (size_t)nt * 16 * DMODEL + kb, DMODEL, lane);
      c[nt] = wmma_f16(a, b, c[nt]);
    }
  }

  int hi = lane >> 4;
  int nl = lane & 15;
#pragma unroll
  for (int nt = 0; nt < 4; ++nt) {
    int n = nq * 64 + nt * 16 + nl;
    float bias = bo[n];
#pragma unroll
    for (int r = 0; r < 8; ++r) {
      int row = mt * 16 + r + 8 * hi;
      out[(size_t)row * DMODEL + n] = c[nt][r] + bias;
    }
  }
}

extern "C" void kernel_launch(void* const* d_in, const int* in_sizes, int n_in,
                              void* d_out, int out_size, void* d_ws, size_t ws_size,
                              hipStream_t stream) {
  const float* inq = (const float*)d_in[0];
  const float* ink = (const float*)d_in[1];
  const float* inv_ = (const float*)d_in[2];
  const float* Wq = (const float*)d_in[3];
  const float* bq = (const float*)d_in[4];
  const float* Wk = (const float*)d_in[5];
  const float* bk = (const float*)d_in[6];
  const float* Wv = (const float*)d_in[7];
  const float* bv = (const float*)d_in[8];
  const float* Wo = (const float*)d_in[9];
  const float* bo = (const float*)d_in[10];
  float* out = (float*)d_out;

  // workspace layout (bytes): 4 x 2MB weight f16 + 4 x 16MB activations = 72 MB
  char* ws = (char*)d_ws;
  const size_t WSZ = (size_t)1024 * 1024 * sizeof(_Float16);             // 2 MB
  const size_t ASZ = (size_t)BATCH * NH * SEQ * HDIM * sizeof(_Float16); // 16 MB
  _Float16* Wqt = (_Float16*)(ws + 0 * WSZ);
  _Float16* Wkt = (_Float16*)(ws + 1 * WSZ);
  _Float16* Wvt = (_Float16*)(ws + 2 * WSZ);
  _Float16* Wot = (_Float16*)(ws + 3 * WSZ);
  _Float16* Qh  = (_Float16*)(ws + 4 * WSZ);
  _Float16* Kh  = (_Float16*)(ws + 4 * WSZ + 1 * ASZ);
  _Float16* Vth = (_Float16*)(ws + 4 * WSZ + 2 * ASZ);
  _Float16* Xh  = (_Float16*)(ws + 4 * WSZ + 3 * ASZ);

  // 1) weights -> f16, transposed to N-major
  wconv_kernel<<<dim3(4096, 4), 256, 0, stream>>>(Wq, Wk, Wv, Wo, Wqt, Wkt, Wvt, Wot);

  // 2) QKV projections: 512 Mtiles x 16 Nquads, 3 matrices
  qkv_proj_kernel<<<dim3(512 * 16, 3), 32, 0, stream>>>(
      inq, ink, inv_, Wqt, Wkt, Wvt, bq, bk, bv, Qh, Kh, Vth);

  // 3) attention: 64 (b,h) x 64 query groups (32 queries each)
  attn_kernel<<<dim3(64 * 64), 32, 0, stream>>>(Qh, Kh, Vth, Xh);

  // 4) output projection
  outproj_kernel<<<dim3(512 * 16), 32, 0, stream>>>(Xh, Wot, bo, out);
}
